// FNO1D_66176856096799
// MI455X (gfx1250) — matrix operand order repaired and checked by reading, not compiled, not run on hardware
//
#include <hip/hip_runtime.h>
#include <cmath>

typedef __attribute__((ext_vector_type(16))) _Float16 v16h;
typedef __attribute__((ext_vector_type(8)))  _Float16 v8h;
typedef __attribute__((ext_vector_type(2)))  _Float16 v2h;
typedef __attribute__((ext_vector_type(8)))  float    v8f;

#define B_  32
#define H_  64
#define L_  8192
#define M_  16
#define NL_ 4

// padded column stride for the transposed LDS tile:
// 72 halves = 144 B = 9*16 B -> every v8h fragment read stays 16B-aligned,
// and the bank stride (36 mod 64) avoids the 2-way conflicts a 64-half
// (128 B, 32-bank) stride would cause.
#define TSTRIDE 72

union V16 { v16h v; v8h h2[2]; _Float16 e[16]; };

__device__ __forceinline__ float silu_f(float x){ return x / (1.0f + __expf(-x)); }

// ---------------------------------------------------------------------------
// K0: precompute DFT bases.
// fb [32][L] : c<16 -> cos(2pi*m*n/L), c>=16 -> -sin   (forward transform)
// ib [L][32] : k<16 -> cos, k>=16 -> +sin              (inverse, sign folded in Yb)
// Phase reduced exactly via (m*n) mod L so f32 trig stays accurate.
// ---------------------------------------------------------------------------
__global__ void basis_kernel(_Float16* __restrict__ fb, _Float16* __restrict__ ib){
  int idx = blockIdx.x*blockDim.x + threadIdx.x;   // L*32 threads
  int n = idx >> 5, c = idx & 31;
  int m = c & 15;
  int r = (m * n) & (L_ - 1);
  float ang = 6.2831853071795864f * (float)r * (1.0f/(float)L_);
  float cv = __cosf(ang), sv = __sinf(ang);
  fb[c*L_ + n]  = (_Float16)((c < 16) ? cv : -sv);
  ib[n*32 + c]  = (_Float16)((c < 16) ? cv :  sv);
}

// ---------------------------------------------------------------------------
// K1: lift (concat with linspace grid) + input projection -> hidden f16 [B][H][L]
// ---------------------------------------------------------------------------
__global__ void lift_kernel(const float* __restrict__ states,
                            const float* __restrict__ pw,
                            const float* __restrict__ pb,
                            _Float16* __restrict__ h16){
  int idx = blockIdx.x*blockDim.x + threadIdx.x;   // B*H*L
  int n = idx & (L_-1);
  int h = (idx >> 13) & (H_-1);
  int b = idx >> 19;
  float x = states[(b << 13) + n];
  float g = (float)n * (1.0f/(float)(L_-1));
  float v = pw[h*2+0]*x + pw[h*2+1]*g + pb[h];
  h16[idx] = (_Float16)v;
}

// ---------------------------------------------------------------------------
// K2: truncated forward DFT as WMMA GEMM.
// [16 rows of (b*H+i)] x [K=8192] x [32 cols (cos | -sin)] per workgroup.
// 8 waves split K (1024 each), partials reduced through LDS.
// ---------------------------------------------------------------------------
__global__ void dft_kernel(const _Float16* __restrict__ h16,
                           const _Float16* __restrict__ fb,
                           float* __restrict__ Xre, float* __restrict__ Xim){
  __shared__ float red[8][2][256];
  const int tid  = threadIdx.x;
  const int lane = tid & 31, wid = tid >> 5;
  const int l15  = lane & 15, lhi = lane >> 4;
  const int rowbase = blockIdx.x << 4;              // rows stay within one b (H=64)
  const _Float16* rowp = h16 + (size_t)(rowbase + l15) * L_;
  const _Float16* fb0  = fb  + (size_t)l15 * L_;         // cos column
  const _Float16* fb1  = fb  + (size_t)(16 + l15) * L_;  // -sin column
  v8f acc0 = {}; v8f acc1 = {};
  const int k0 = wid << 10;
  for (int k = k0; k < k0 + 1024; k += 32) {
    __builtin_prefetch(rowp + k + 256, 0, 1);       // global_prefetch_b8
    V16 a, b0, b1;
    // A frag (16x32 f16): elements = 16 contiguous K starting at k+16*lhi
    a.h2[0] = *(const v8h*)(rowp + k + 16*lhi);
    a.h2[1] = *(const v8h*)(rowp + k + 16*lhi + 8);
    // B frag (32x16 f16): elements = 16 contiguous K starting at k+16*lhi
    b0.v = *(const v16h*)(fb0 + k + 16*lhi);
    b1.v = *(const v16h*)(fb1 + k + 16*lhi);
    acc0 = __builtin_amdgcn_wmma_f32_16x16x32_f16(false, a.v, false, b0.v,
                                                  (short)0, acc0, false, false);
    acc1 = __builtin_amdgcn_wmma_f32_16x16x32_f16(false, a.v, false, b1.v,
                                                  (short)0, acc1, false, false);
  }
#pragma unroll
  for (int r = 0; r < 8; ++r) {
    red[wid][0][r*32 + lane] = acc0[r];
    red[wid][1][r*32 + lane] = acc1[r];
  }
  __syncthreads();
  if (wid < 2) {
    float* out = (wid == 0) ? Xre : Xim;
#pragma unroll
    for (int r = 0; r < 8; ++r) {
      float s = 0.f;
#pragma unroll
      for (int w = 0; w < 8; ++w) s += red[w][wid][r*32 + lane];
      int row = rowbase + r + 8*lhi;                // C layout: M = r / r+8
      out[row*M_ + l15] = s;
    }
  }
}

// ---------------------------------------------------------------------------
// K3: per-mode complex mixing Y[b,o,m] = sum_i X[b,i,m] * W[i,o,m],
// with irfft scaling folded in: Ya = Re(Y)*(m==0?1:2)/L, Yb = -Im(Y)*(...)/L
// ---------------------------------------------------------------------------
__global__ void mix_kernel(const float* __restrict__ Xre, const float* __restrict__ Xim,
                           const float* __restrict__ Wre, const float* __restrict__ Wim,
                           float* __restrict__ Ya, float* __restrict__ Yb){
  int idx = blockIdx.x*blockDim.x + threadIdx.x;    // B*H*M = b*1024 + o*16 + m
  int m = idx & 15, o = (idx >> 4) & 63, b = idx >> 10;
  float yr = 0.f, yi = 0.f;
  for (int i = 0; i < H_; ++i){
    float xr = Xre[((b << 6) + i)*M_ + m];
    float xi = Xim[((b << 6) + i)*M_ + m];
    float wr = Wre[(i*H_ + o)*M_ + m];
    float wi = Wim[(i*H_ + o)*M_ + m];
    yr = fmaf(xr, wr, yr); yr = fmaf(-xi, wi, yr);
    yi = fmaf(xr, wi, yi); yi = fmaf( xi, wr, yi);
  }
  float sc = ((m == 0) ? 1.0f : 2.0f) * (1.0f/(float)L_);
  Ya[idx] =  yr * sc;
  Yb[idx] = -yi * sc;
}

// ---------------------------------------------------------------------------
// K4: fused (inverse-transform + pointwise conv + bias + SiLU).
// Out[o, n] tile (16x16) = W[64x64] * hidden (2 WMMA, K=64)
//                        + Y[64x32] * ibasis (1 WMMA, K=32, if use_spec)
// One workgroup: fixed b, 256 columns of n (8 sub-tiles of 32), all 64 o.
// Hidden tile staged into a double-buffered, padded, transposed LDS tile
// ([col][k], stride TSTRIDE) so every B-fragment is two aligned b128 LDS reads.
// ---------------------------------------------------------------------------
__global__ void layer_kernel(const _Float16* __restrict__ hin,
                             _Float16* __restrict__ hout,
                             const _Float16* __restrict__ ib,
                             const float* __restrict__ Ya, const float* __restrict__ Yb,
                             const float* __restrict__ W, const float* __restrict__ bias,
                             int use_spec){
  __shared__ _Float16 lds_t[2][32*TSTRIDE];         // [buf][col][k], padded
  const int tid  = threadIdx.x;
  const int lane = tid & 31, wid = tid >> 5;
  const int l15  = lane & 15, lhi = lane >> 4;
  const int b      = blockIdx.x >> 5;
  const int nbase0 = (blockIdx.x & 31) << 8;
  const int obase  = (wid >> 1) << 4;
  const int nsub   = (wid & 1) << 4;
  const int arow   = obase + l15;

  V16 aw0, aw1, ay;
#pragma unroll
  for (int e = 0; e < 16; ++e){
    int k = e + 16*lhi;                             // A-matrix K pattern
    aw0.e[e] = (_Float16)W[arow*H_ + k];
    aw1.e[e] = (_Float16)W[arow*H_ + 32 + k];
  }
  if (use_spec){
#pragma unroll
    for (int e = 0; e < 16; ++e){
      int k = e + 16*lhi;
      float v = (k < 16) ? Ya[((b << 6) + arow)*M_ + k]
                         : Yb[((b << 6) + arow)*M_ + (k - 16)];
      ay.e[e] = (_Float16)v;
    }
  } else {
#pragma unroll
    for (int e = 0; e < 16; ++e) ay.e[e] = (_Float16)0.f;
  }

  // staging index: 16 threads sweep one k-row (32 cols) -> coalesced v2h loads
  const int scp = tid & 15;          // column pair 0..15 -> cols 2cp, 2cp+1
  const int sk0 = tid >> 4;          // k rows: sk0, sk0+16, sk0+32, sk0+48

  for (int jj = 0; jj < 8; ++jj){
    const int n0 = nbase0 + (jj << 5);
    _Float16* buf = lds_t[jj & 1];
#pragma unroll
    for (int s = 0; s < 4; ++s){
      int k = sk0 + 16*s;
      v2h hv = *(const v2h*)(hin + (size_t)((b << 6) + k)*L_ + n0 + 2*scp);
      buf[(2*scp  )*TSTRIDE + k] = hv[0];           // transpose into LDS
      buf[(2*scp+1)*TSTRIDE + k] = hv[1];
    }
    __syncthreads();                                // single barrier (dbl-buffered)

    v8f acc = {};
    const int colL = nsub + l15;
    const _Float16* cp = buf + colL*TSTRIDE;
    V16 b0, b1;
    b0.h2[0] = *(const v8h*)(cp + 16*lhi);
    b0.h2[1] = *(const v8h*)(cp + 16*lhi + 8);
    b1.h2[0] = *(const v8h*)(cp + 32 + 16*lhi);
    b1.h2[1] = *(const v8h*)(cp + 32 + 16*lhi + 8);
    acc = __builtin_amdgcn_wmma_f32_16x16x32_f16(false, aw0.v, false, b0.v,
                                                 (short)0, acc, false, false);
    acc = __builtin_amdgcn_wmma_f32_16x16x32_f16(false, aw1.v, false, b1.v,
                                                 (short)0, acc, false, false);
    if (use_spec){
      V16 bi;
      bi.v = *(const v16h*)(ib + (size_t)(n0 + colL)*32 + 16*lhi);
      acc = __builtin_amdgcn_wmma_f32_16x16x32_f16(false, ay.v, false, bi.v,
                                                   (short)0, acc, false, false);
    }
#pragma unroll
    for (int r = 0; r < 8; ++r){
      int o   = obase + r + 8*lhi;
      int col = n0 + colL;
      float v = acc[r] + bias[o];
      v = silu_f(v);
      hout[(size_t)((b << 6) + o)*L_ + col] = (_Float16)v;
    }
  }
}

// ---------------------------------------------------------------------------
// K6: output projection (H -> 1) + residual add.
// ---------------------------------------------------------------------------
__global__ void out_kernel(const _Float16* __restrict__ hin,
                           const float* __restrict__ states,
                           const float* __restrict__ w2, const float* __restrict__ b2,
                           float* __restrict__ out){
  int idx = blockIdx.x*blockDim.x + threadIdx.x;    // B*L
  int b = idx >> 13, n = idx & (L_-1);
  float s = b2[0];
#pragma unroll 8
  for (int i = 0; i < H_; ++i)
    s = fmaf(w2[i], (float)hin[(size_t)((b << 6) + i)*L_ + n], s);
  out[idx] = states[idx] + s;
}

// ---------------------------------------------------------------------------
extern "C" void kernel_launch(void* const* d_in, const int* in_sizes, int n_in,
                              void* d_out, int out_size, void* d_ws, size_t ws_size,
                              hipStream_t stream){
  const float* states  = (const float*)d_in[0];
  const float* proj_w  = (const float*)d_in[1];
  const float* proj_b  = (const float*)d_in[2];
  const float* spec_re = (const float*)d_in[3];
  const float* spec_im = (const float*)d_in[4];
  const float* pw_w    = (const float*)d_in[5];
  const float* pw_b    = (const float*)d_in[6];
  const float* op1_w   = (const float*)d_in[7];
  const float* op1_b   = (const float*)d_in[8];
  const float* op2_w   = (const float*)d_in[9];
  const float* op2_b   = (const float*)d_in[10];
  float* out = (float*)d_out;

  // workspace layout (bytes)
  char* ws = (char*)d_ws;
  _Float16* fb  = (_Float16*)(ws + 0);          //  32*L*2   = 512 KB
  _Float16* ib  = (_Float16*)(ws + 524288);     //  L*32*2   = 512 KB
  _Float16* hA  = (_Float16*)(ws + 1048576);    //  B*H*L*2  =  32 MB
  _Float16* hB  = (_Float16*)(ws + 34603008);   //  B*H*L*2  =  32 MB
  float*    Xre = (float*)   (ws + 68157440);   //  B*H*M*4  = 128 KB
  float*    Xim = (float*)   (ws + 68288512);
  float*    Ya  = (float*)   (ws + 68419584);
  float*    Yb  = (float*)   (ws + 68550656);   // end ~68.7 MB

  basis_kernel<<<(L_*32)/256, 256, 0, stream>>>(fb, ib);
  lift_kernel<<<(B_*H_*L_)/256, 256, 0, stream>>>(states, proj_w, proj_b, hA);

  _Float16* cur = hA;
  _Float16* nxt = hB;
  for (int l = 0; l < NL_; ++l){
    dft_kernel<<<(B_*H_)/16, 256, 0, stream>>>(cur, fb, Xre, Xim);
    mix_kernel<<<(B_*H_*M_)/256, 256, 0, stream>>>(Xre, Xim,
        spec_re + (size_t)l*H_*H_*M_, spec_im + (size_t)l*H_*H_*M_, Ya, Yb);
    layer_kernel<<<B_*32, 256, 0, stream>>>(cur, nxt, ib, Ya, Yb,
        pw_w + (size_t)l*H_*H_, pw_b + (size_t)l*H_, 1);
    _Float16* t = cur; cur = nxt; nxt = t;
  }
  // op1 + SiLU (pure pointwise, spectral path disabled)
  layer_kernel<<<B_*32, 256, 0, stream>>>(cur, nxt, ib, nullptr, nullptr,
                                          op1_w, op1_b, 0);
  { _Float16* t = cur; cur = nxt; nxt = t; }
  out_kernel<<<(B_*L_)/256, 256, 0, stream>>>(cur, states, op2_w, op2_b, out);
}
